// MultiHeadAttention_36721970380862
// MI455X (gfx1250) — matrix operand order repaired and checked by reading, not compiled, not run on hardware
//
#include <hip/hip_runtime.h>

// ---------------------------------------------------------------------------
// Multi-head causal attention forward for MI455X (gfx1250, wave32, WMMA).
// bf16 WMMA everywhere (f32 accum). GEMM A-tiles staged by the Tensor Data
// Mover (TENSORcnt), B-tiles by GLOBAL_LOAD_ASYNC_TO_LDS_B128 (ASYNCcnt),
// double-buffered LDS. V pre-transposed so P@V fragments are b128 loads.
//   x -> xbf; w_qkv -> wqkvT [N][K]; w_out -> woutT [N][K]
//   qkv  = xbf @ wqkvT^T + b_qkv                  (bf16, WMMA 128x128 tiles)
//   vt   = transpose of V slice of qkv            (bf16, [b*h][64][2048])
//   attn = causal flash attention (h=16, d=64)    (bf16, WMMA)
//   out  = attn @ woutT^T + b_out                 (fp32, WMMA)
// Workspace (bf16): 8M+3M+1M+24M+8M+8M elems = 104 MB.
// ---------------------------------------------------------------------------

typedef __attribute__((ext_vector_type(16))) __bf16        v16bf;
typedef __attribute__((ext_vector_type(8)))  float         v8f;
typedef __attribute__((ext_vector_type(4)))  int           v4i;
typedef __attribute__((ext_vector_type(4)))  unsigned int  u32x4;
typedef __attribute__((ext_vector_type(8)))  int           i32x8;
typedef __attribute__((ext_vector_type(4)))  int           i32x4;

union Frag {
  v16bf    v;
  uint4    u[2];
  unsigned u32[8];
};

static __device__ __forceinline__ unsigned short f2bf_u16(float f) {
  union { float f; unsigned u; } c; c.f = f;
  unsigned r = c.u + 0x7FFFu + ((c.u >> 16) & 1u);   // round-to-nearest-even
  return (unsigned short)(r >> 16);
}
static __device__ __forceinline__ unsigned pack2(float a, float b) {
  return (unsigned)f2bf_u16(a) | ((unsigned)f2bf_u16(b) << 16);
}
static __device__ __forceinline__ uint4 cvt8(const float* p) {
  float4 f0 = *(const float4*)p;
  float4 f1 = *(const float4*)(p + 4);
  uint4 r;
  r.x = pack2(f0.x, f0.y);
  r.y = pack2(f0.z, f0.w);
  r.z = pack2(f1.x, f1.y);
  r.w = pack2(f1.z, f1.w);
  return r;
}
static __device__ __forceinline__ v8f wmma_bf16(v16bf a, v16bf b, v8f c) {
  return __builtin_amdgcn_wmma_f32_16x16x32_bf16(false, a, false, b, (short)0, c,
                                                 false, false);
}

// --- CDNA5 async global->LDS staging (ASYNCcnt) ----------------------------
#if __has_builtin(__builtin_amdgcn_global_load_async_to_lds_b128) && \
    __has_builtin(__builtin_amdgcn_s_wait_asynccnt)
#define HAVE_ASYNC_LDS 1
static __device__ __forceinline__ void copy16(unsigned short* lds_dst,
                                              const unsigned short* gsrc) {
  __builtin_amdgcn_global_load_async_to_lds_b128(
      (__attribute__((address_space(1))) v4i*)(uintptr_t)gsrc,
      (__attribute__((address_space(3))) v4i*)(unsigned)(uintptr_t)lds_dst,
      0, 0);
}
static __device__ __forceinline__ void staging_wait() {
  __builtin_amdgcn_s_wait_asynccnt(0);
}
#else
#define HAVE_ASYNC_LDS 0
static __device__ __forceinline__ void copy16(unsigned short* lds_dst,
                                              const unsigned short* gsrc) {
  *(uint4*)lds_dst = *(const uint4*)gsrc;
}
static __device__ __forceinline__ void staging_wait() {}
#endif

// --- CDNA5 Tensor Data Mover (TENSORcnt) -----------------------------------
// Builds a D# for a 2D bf16 tile: tile_dim0=32 elems (K), tile_dim1=128 rows,
// row stride = K elems, packed contiguously at lds_addr (row-major 128x32).
#if __has_builtin(__builtin_amdgcn_tensor_load_to_lds) && \
    __has_builtin(__builtin_amdgcn_s_wait_tensorcnt)
#define HAVE_TDM 1
static __device__ __forceinline__ void tdm_load_tile_128x32(
    const unsigned short* gtile, unsigned short* lds, int row_stride_elems) {
  unsigned long long ga = (unsigned long long)(uintptr_t)gtile;
  u32x4 g0;
  g0[0] = 1u;                                    // count=1 (valid), user mode
  g0[1] = (unsigned)(uintptr_t)lds;              // lds_addr (bytes)
  g0[2] = (unsigned)ga;                          // global_addr[31:0]
  g0[3] = (unsigned)(ga >> 32) | (2u << 30);     // global_addr[56:32] | type=2
  i32x8 g1;
  g1[0] = 0x10000;                               // data_size=1 (2 bytes)
  g1[1] = (row_stride_elems & 0xFFFF) << 16;     // tensor_dim0[15:0]
  g1[2] = ((unsigned)row_stride_elems >> 16) |   // tensor_dim0[31:16]
          (8192 << 16);                          // tensor_dim1[15:0] = 8192
  g1[3] = 0 | (32 << 16);                        // tensor_dim1[31:16]=0 | tile_dim0=32
  g1[4] = 128;                                   // tile_dim1=128, tile_dim2=0
  g1[5] = row_stride_elems;                      // tensor_dim0_stride[31:0]
  g1[6] = 0;                                     // stride hi, dim1_stride lo
  g1[7] = 0;
  i32x4 gz = {0, 0, 0, 0};
#if defined(__clang_major__) && __clang_major__ >= 23
  i32x8 gz8 = {0, 0, 0, 0, 0, 0, 0, 0};
  __builtin_amdgcn_tensor_load_to_lds(g0, g1, gz, gz, gz8, 0);
#else
  __builtin_amdgcn_tensor_load_to_lds(g0, g1, gz, gz, 0);
#endif
}
#else
#define HAVE_TDM 0
#endif

#define SEQ    2048
#define QKVLD  3072
#define DMODEL 1024
#define DHEAD  64

// ---------------------------------------------------------------------------
// Conversion / transpose passes (one-time, bandwidth-trivial).
// ---------------------------------------------------------------------------
__global__ __launch_bounds__(256) void cvt_f32_bf16(
    const float* __restrict__ src, unsigned short* __restrict__ dst, int n) {
  int i = ((int)blockIdx.x * 256 + (int)threadIdx.x) * 8;
  if (i + 8 <= n) *(uint4*)(dst + i) = cvt8(src + i);
}

// w[K][N] fp32 -> wt[N][K] bf16
__global__ __launch_bounds__(256) void cvt_transpose_bf16(
    const float* __restrict__ w, unsigned short* __restrict__ wt, int K, int N) {
  int n = (int)blockIdx.x * 32 + (int)threadIdx.x % 32;
  int k = (int)blockIdx.y * 8 + (int)threadIdx.x / 32;
  wt[(size_t)n * K + k] = f2bf_u16(w[(size_t)k * N + n]);
}

// V slice of qkv -> vt[b*h][64][2048] (coalesced both sides via LDS tile)
__global__ __launch_bounds__(256) void transpose_v_bf16(
    const unsigned short* __restrict__ qkv, unsigned short* __restrict__ vt) {
  __shared__ __align__(16) unsigned short tile[32][72];   // 32 tok x 64 d (+pad)
  const int bh    = (int)blockIdx.y;                      // 0..63
  const int batch = bh >> 4, head = bh & 15;
  const int tok0  = (int)blockIdx.x * 32;
  {
    int tok = (int)threadIdx.x >> 3;
    int c8  = ((int)threadIdx.x & 7) * 8;
    const unsigned short* src = qkv + (size_t)(batch * SEQ + tok0 + tok) * QKVLD +
                                2 * DMODEL + head * DHEAD + c8;
    *(uint4*)&tile[tok][c8] = *(const uint4*)src;
  }
  __syncthreads();
  {
    int d  = (int)threadIdx.x >> 2;
    int tk = ((int)threadIdx.x & 3) * 8;
    uint4 o;
    unsigned short* po = (unsigned short*)&o;
#pragma unroll
    for (int i = 0; i < 8; ++i) po[i] = tile[tk + i][d];
    *(uint4*)(vt + ((size_t)bh * DHEAD + d) * SEQ + tok0 + tk) = o;
  }
}

// ---------------------------------------------------------------------------
// GEMM: C[M,N] = A[M,K] @ Bt[N,K]^T + bias[N], bf16 in, WMMA bf16.
// Block tile 128x128, K-step 32, 8 waves; each wave = 32x64 slab = 8 WMMA
// per K-step. A staged by TDM (wave 0 issues descriptor), B by per-thread
// async-LDS copies; double-buffered.
// ---------------------------------------------------------------------------
#define BM 128
#define BN 128
#define BK 32

template <bool OUT_BF16>
__global__ __launch_bounds__(256) void gemm_bf16_wmma(
    const unsigned short* __restrict__ A,    // [M,K] bf16
    const unsigned short* __restrict__ Bt,   // [N,K] bf16 (pre-transposed)
    const float* __restrict__ bias,          // [N] fp32
    void* __restrict__ Cout,                 // [M,N] bf16 or fp32
    int M, int N, int K)
{
  __shared__ __align__(16) unsigned short lsA[2][BM][BK];   // 2 x 8KB
  __shared__ __align__(16) unsigned short lsB[2][BN][BK];   // 2 x 8KB

  const int tid  = threadIdx.x;
  const int wave = tid >> 5;
  const int lane = tid & 31;
  const int half = lane >> 4;
  const int l16  = lane & 15;

  const int bm0 = (int)blockIdx.y * BM;
  const int bn0 = (int)blockIdx.x * BN;
  const int wm  = wave >> 1;            // 0..3 -> 32-row slab
  const int wn  = wave & 1;             // 0..1 -> 64-col slab

  v8f acc[2][4] = {};                   // [mi][ni]

  auto stage = [&](int buf, int k0) {
#if HAVE_TDM
    if (wave == 0)                      // one TDM descriptor moves the A tile
      tdm_load_tile_128x32(A + (size_t)bm0 * K + k0, &lsA[buf][0][0], K);
#endif
#pragma unroll
    for (int j = 0; j < 2; ++j) {
      int c = tid * 2 + j;
      int r = c >> 2, c8 = (c & 3) * 8;
#if !HAVE_TDM
      copy16(&lsA[buf][r][c8], A + (size_t)(bm0 + r) * K + k0 + c8);
#endif
      copy16(&lsB[buf][r][c8], Bt + (size_t)(bn0 + r) * K + k0 + c8);
    }
  };

  stage(0, 0);
  const int nk = K / BK;
  for (int ki = 0; ki < nk; ++ki) {
    const int buf = ki & 1;
    staging_wait();                     // own ASYNCcnt drained
#if HAVE_TDM
    if (wave == 0) __builtin_amdgcn_s_wait_tensorcnt(0);
#endif
    __syncthreads();                    // all stores visible / prior reads done
    if (ki + 1 < nk) stage(buf ^ 1, (ki + 1) * BK);

    Frag af[2], bf[4];
#pragma unroll
    for (int mi = 0; mi < 2; ++mi) {
      int m = wm * 32 + mi * 16 + l16;
      af[mi].u[0] = *(const uint4*)&lsA[buf][m][half * 8];
      af[mi].u[1] = *(const uint4*)&lsA[buf][m][16 + half * 8];
    }
#pragma unroll
    for (int ni = 0; ni < 4; ++ni) {
      const uint4* p = (const uint4*)&lsB[buf][wn * 64 + ni * 16 + l16][half * 8];
      bf[ni].u[0] = p[0];
      bf[ni].u[1] = p[2];
    }
#pragma unroll
    for (int mi = 0; mi < 2; ++mi)
#pragma unroll
      for (int ni = 0; ni < 4; ++ni)
        acc[mi][ni] = wmma_bf16(af[mi].v, bf[ni].v, acc[mi][ni]);
  }

  // Epilogue: lane -> rows wm*32+mi*16+half*8+r, cols wn*64+ni*16+l16
#pragma unroll
  for (int ni = 0; ni < 4; ++ni) {
    int col = bn0 + wn * 64 + ni * 16 + l16;
    float bv = bias[col];
#pragma unroll
    for (int mi = 0; mi < 2; ++mi) {
      int row0 = bm0 + wm * 32 + mi * 16 + half * 8;
#pragma unroll
      for (int r = 0; r < 8; ++r) {
        float v = acc[mi][ni][r] + bv;
        if (OUT_BF16)
          ((unsigned short*)Cout)[(size_t)(row0 + r) * N + col] = f2bf_u16(v);
        else
          ((float*)Cout)[(size_t)(row0 + r) * N + col] = v;
      }
    }
  }
}

// ---------------------------------------------------------------------------
// Flash attention, causal, h=16, d=64, n=2048, bf16 in/out.
// One wave per (batch, head, 16-row Q tile); 8 independent waves per block.
// ---------------------------------------------------------------------------
__global__ __launch_bounds__(256) void mha_flash_attn_bf16(
    const unsigned short* __restrict__ qkv,   // [b*n,3072] bf16
    const unsigned short* __restrict__ vt,    // [b*h][64][2048] bf16
    unsigned short* __restrict__ attn_out)    // [b*n,1024] bf16
{
  __shared__ __align__(16) unsigned short lsP[8][16 * 32];   // per-wave P tile

  const int tid  = threadIdx.x;
  const int wave = tid >> 5;
  const int lane = tid & 31;
  const int half = lane >> 4;
  const int l16  = lane & 15;

  const int gw    = (int)blockIdx.x * 8 + wave;   // 8192 wave tasks
  const int qt    = gw & 127;
  const int head  = (gw >> 7) & 15;
  const int batch = gw >> 11;

  const int   qrow0 = qt * 16;
  const float scale = 0.125f;                     // 1/sqrt(64)
  const unsigned short* base  = qkv + (size_t)batch * SEQ * QKVLD;
  const unsigned short* vbase = vt + (size_t)(batch * 16 + head) * DHEAD * SEQ;

  // --- Q fragments (d-chunks [0,32) and [32,64)) --------------------------
  Frag qf0, qf1;
  {
    const unsigned short* qr = base + (size_t)(qrow0 + l16) * QKVLD + head * DHEAD;
    qf0.u[0] = *(const uint4*)(qr + half * 8);
    qf0.u[1] = *(const uint4*)(qr + 16 + half * 8);
    qf1.u[0] = *(const uint4*)(qr + 32 + half * 8);
    qf1.u[1] = *(const uint4*)(qr + 48 + half * 8);
  }

  v8f   o0 = {}, o1 = {}, o2 = {}, o3 = {};
  float mrow[8], lrow[8];
#pragma unroll
  for (int r = 0; r < 8; ++r) { mrow[r] = -__builtin_inff(); lrow[r] = 0.0f; }

  for (int j0 = 0; j0 < qrow0 + 16; j0 += 32) {
    // prefetch next key block: K rows (one per lane) + vt rows
    if (j0 + 32 < qrow0 + 16) {
      __builtin_prefetch(base + (size_t)(j0 + 32 + lane) * QKVLD +
                         DMODEL + head * DHEAD, 0, 1);
      __builtin_prefetch(vbase + (size_t)(lane * 2) * SEQ + j0 + 32, 0, 1);
    }

    // --- S = Q K^T : two 16-key tiles x two d-chunks ----------------------
    v8f s0 = {}, s1 = {};
#pragma unroll
    for (int c = 0; c < 2; ++c) {
      const unsigned short* kr0 = base + (size_t)(j0 + l16) * QKVLD +
                                  DMODEL + head * DHEAD + c * 32 + half * 8;
      Frag kb0, kb1;
      kb0.u[0] = *(const uint4*)(kr0);
      kb0.u[1] = *(const uint4*)(kr0 + 16);
      const unsigned short* kr1 = kr0 + (size_t)16 * QKVLD;
      kb1.u[0] = *(const uint4*)(kr1);
      kb1.u[1] = *(const uint4*)(kr1 + 16);
      s0 = wmma_bf16((c == 0 ? qf0.v : qf1.v), kb0.v, s0);
      s1 = wmma_bf16((c == 0 ? qf0.v : qf1.v), kb1.v, s1);
    }

    // --- online softmax ---------------------------------------------------
    const int col0 = j0 + l16;
    const int col1 = j0 + 16 + l16;
    float alpha[8];
#pragma unroll
    for (int r = 0; r < 8; ++r) {
      int rowr = qrow0 + half * 8 + r;
      float v0 = (col0 <= rowr) ? s0[r] * scale : -__builtin_inff();
      float v1 = (col1 <= rowr) ? s1[r] * scale : -__builtin_inff();
      float mx = fmaxf(v0, v1);
#pragma unroll
      for (int off = 1; off < 16; off <<= 1)
        mx = fmaxf(mx, __shfl_xor(mx, off, 16));
      float mn = fmaxf(mrow[r], mx);
      float p0 = __expf(v0 - mn);
      float p1 = __expf(v1 - mn);
      float ps = p0 + p1;
#pragma unroll
      for (int off = 1; off < 16; off <<= 1)
        ps += __shfl_xor(ps, off, 16);
      alpha[r] = __expf(mrow[r] - mn);
      mrow[r]  = mn;
      lrow[r]  = lrow[r] * alpha[r] + ps;
      lsP[wave][(half * 8 + r) * 32 + l16]      = f2bf_u16(p0);
      lsP[wave][(half * 8 + r) * 32 + 16 + l16] = f2bf_u16(p1);
    }
#pragma unroll
    for (int r = 0; r < 8; ++r) {
      o0[r] *= alpha[r]; o1[r] *= alpha[r];
      o2[r] *= alpha[r]; o3[r] *= alpha[r];
    }

    // --- reload P as A fragment (same-wave LDS: DScnt-ordered) ------------
    Frag pf;
    {
      const unsigned short* p = &lsP[wave][l16 * 32];
      pf.u[0] = *(const uint4*)(p + half * 8);
      pf.u[1] = *(const uint4*)(p + 16 + half * 8);
    }

    // --- O += P V : four 16-wide d-groups, contiguous vt rows -------------
#pragma unroll
    for (int g = 0; g < 4; ++g) {
      const unsigned short* vp = vbase + (size_t)(g * 16 + l16) * SEQ +
                                 j0 + half * 8;
      Frag vb;
      vb.u[0] = *(const uint4*)(vp);
      vb.u[1] = *(const uint4*)(vp + 16);
      v8f& og = (g == 0) ? o0 : (g == 1) ? o1 : (g == 2) ? o2 : o3;
      og = wmma_bf16(pf.v, vb.v, og);
    }
  }

  // --- normalize + store (bf16) -------------------------------------------
#pragma unroll
  for (int r = 0; r < 8; ++r) {
    float inv = 1.0f / lrow[r];
    int row = qrow0 + half * 8 + r;
    unsigned short* dst = attn_out + (size_t)(batch * SEQ + row) * DMODEL +
                          head * DHEAD + l16;
    dst[0]  = f2bf_u16(o0[r] * inv);
    dst[16] = f2bf_u16(o1[r] * inv);
    dst[32] = f2bf_u16(o2[r] * inv);
    dst[48] = f2bf_u16(o3[r] * inv);
  }
}

// ---------------------------------------------------------------------------
extern "C" void kernel_launch(void* const* d_in, const int* in_sizes, int n_in,
                              void* d_out, int out_size, void* d_ws, size_t ws_size,
                              hipStream_t stream) {
  const float* x     = (const float*)d_in[0];   // [4,2048,1024]
  const float* w_qkv = (const float*)d_in[1];   // [1024,3072]
  const float* b_qkv = (const float*)d_in[2];   // [3072]
  const float* w_out = (const float*)d_in[3];   // [1024,1024]
  const float* b_out = (const float*)d_in[4];   // [1024]
  float* out = (float*)d_out;                   // [4,2048,1024]

  const int Mtok = 4 * SEQ;                     // 8192 token rows

  // Workspace (bf16): xbf | wqkvT | woutT | qkv | attn | vt  = 104 MB
  unsigned short* xbf    = (unsigned short*)d_ws;
  unsigned short* wqkvT  = xbf    + (size_t)Mtok * DMODEL;    // [3072][1024]
  unsigned short* woutT  = wqkvT  + (size_t)QKVLD * DMODEL;   // [1024][1024]
  unsigned short* qkvbf  = woutT  + (size_t)DMODEL * DMODEL;  // [8192][3072]
  unsigned short* attnbf = qkvbf  + (size_t)Mtok * QKVLD;     // [8192][1024]
  unsigned short* vtbf   = attnbf + (size_t)Mtok * DMODEL;    // [64][64][2048]

  dim3 blk(256);

  cvt_f32_bf16<<<dim3((Mtok * DMODEL) / (8 * 256)), blk, 0, stream>>>(
      x, xbf, Mtok * DMODEL);
  cvt_transpose_bf16<<<dim3(QKVLD / 32, DMODEL / 8), blk, 0, stream>>>(
      w_qkv, wqkvT, DMODEL, QKVLD);
  cvt_transpose_bf16<<<dim3(DMODEL / 32, DMODEL / 8), blk, 0, stream>>>(
      w_out, woutT, DMODEL, DMODEL);

  gemm_bf16_wmma<true><<<dim3(QKVLD / BN, Mtok / BM), blk, 0, stream>>>(
      xbf, wqkvT, b_qkv, qkvbf, Mtok, QKVLD, DMODEL);

  transpose_v_bf16<<<dim3(SEQ / 32, 64), blk, 0, stream>>>(qkvbf, vtbf);

  mha_flash_attn_bf16<<<dim3(1024), blk, 0, stream>>>(qkvbf, vtbf, attnbf);

  gemm_bf16_wmma<false><<<dim3(DMODEL / BN, Mtok / BM), blk, 0, stream>>>(
      attnbf, woutT, b_out, out, Mtok, DMODEL, DMODEL);
}
